// upsample_1434519077617
// MI455X (gfx1250) — matrix operand order repaired and checked by reading, not compiled, not run on hardware
//
#include <hip/hip_runtime.h>

// ---------------------------------------------------------------------------
// HaloAttention + 1x1 conv + PixelShuffle for MI455X (gfx1250, wave32, WMMA).
// B=8 C=256 H=W=64, block=8 halo=4 heads=4 d=64, R=16, Nb=512 windows.
// GEMMs on v_wmma_f32_16x16x32_f16; A-tile staging via TDM tensor_load_to_lds
// (double-buffered, s_wait_tensorcnt), sim kept entirely in LDS.
// ---------------------------------------------------------------------------

typedef _Float16 fp16_t;
typedef __attribute__((ext_vector_type(16))) _Float16 v16h;
typedef __attribute__((ext_vector_type(8)))  float    v8f;
typedef __attribute__((ext_vector_type(4)))  unsigned int u32x4;
typedef __attribute__((ext_vector_type(4)))  int      i32x4;
typedef __attribute__((ext_vector_type(8)))  int      i32x8;

#define WMMA_F16(a, b, c) \
  __builtin_amdgcn_wmma_f32_16x16x32_f16(false, (a), false, (b), (short)0, (c), false, false)

// ---------------------------------------------------------------------------
// Tensor Data Mover: 2D tile (tile_x elems wide, tile_y rows) of f16 from
// global (row stride = stride_elems) into LDS, packed row-major.
// D# layout per CDNA5 ISA 8.3/8.4. Issue from one wave; EXEC is ignored.
// ---------------------------------------------------------------------------
__device__ __forceinline__ void tdm_load_2d_f16(const void* gptr, void* lptr,
                                                unsigned tile_x, unsigned tile_y,
                                                unsigned stride_elems) {
  const unsigned long long ga = (unsigned long long)(uintptr_t)gptr;
  u32x4 g0;
  g0[0] = 1u;                                   // count=1, user descriptor
  g0[1] = (unsigned)(uintptr_t)lptr;            // LDS byte address (low 32 bits)
  g0[2] = (unsigned)ga;                         // global_addr[31:0]
  g0[3] = ((unsigned)(ga >> 32) & 0x01FFFFFFu)  // global_addr[56:32]
          | (2u << 30);                         // type = 2 (image)
  i32x8 g1;
  g1[0] = (int)(1u << 16);                      // wg_mask=0, data_size=1 (2B)
  g1[1] = (int)((tile_x & 0xFFFFu) << 16);      // tensor_dim0[15:0]
  g1[2] = (int)((tile_x >> 16) | ((tile_y & 0xFFFFu) << 16));  // dim0 hi | dim1 lo
  g1[3] = (int)((tile_y >> 16) | (tile_x << 16));              // dim1 hi | tile_dim0
  g1[4] = (int)(tile_y & 0xFFFFu);              // tile_dim1 (tile_dim2 = 0)
  g1[5] = (int)stride_elems;                    // tensor_dim0_stride[31:0]
  g1[6] = 0;                                    // stride0[47:32] | dim1_stride lo
  g1[7] = 0;
  const i32x4 z4 = {0, 0, 0, 0};
#if defined(__clang_major__) && __clang_major__ >= 23
  const i32x8 z8 = {0, 0, 0, 0, 0, 0, 0, 0};
  __builtin_amdgcn_tensor_load_to_lds(g0, g1, z4, z4, z8, 0);
#else
  __builtin_amdgcn_tensor_load_to_lds(g0, g1, z4, z4, 0);
#endif
}

// A-matrix 16x32 fragment (ISA 7.12.2): lane holds row m=lane&15;
// lanes 0-15: VGPR0-3 -> K=0..7, VGPR4-7 -> K=16..23; lanes 16-31: +8.
__device__ __forceinline__ v16h load_a_frag(const fp16_t* base, int stride) {
  const int lane = threadIdx.x & 31;
  const int m = lane & 15, hf = lane >> 4;
  v16h f;
#pragma unroll
  for (int j = 0; j < 8; ++j) {
    const int kk = (j < 4) ? (8 * hf + 2 * j) : (16 + 8 * hf + 2 * (j - 4));
    f[2 * j]     = base[m * stride + kk];
    f[2 * j + 1] = base[m * stride + kk + 1];
  }
  return f;
}

// B-matrix 32x16 fragment: lane holds column n=lane&15;
// lanes 0-15: VGPR0-3 -> K=0..7, VGPR4-7 -> K=16..23; lanes 16-31: +8.
__device__ __forceinline__ v16h load_b_frag(const fp16_t* base, int stride) {
  const int lane = threadIdx.x & 31;
  const int n = lane & 15, hf = lane >> 4;
  v16h f;
#pragma unroll
  for (int j = 0; j < 8; ++j) {
    const int kk = (j < 4) ? (8 * hf + 2 * j) : (16 + 8 * hf + 2 * (j - 4));
    f[2 * j]     = base[n * stride + kk];
    f[2 * j + 1] = base[n * stride + kk + 1];
  }
  return f;
}

// ---------------------------------------------------------------------------
// Prep: weight (N rows x K cols, f32) -> B matrix (K x N, f16)
// ---------------------------------------------------------------------------
__global__ void transpose_w_kernel(const float* __restrict__ in,
                                   fp16_t* __restrict__ out, int N, int K) {
  int i = blockIdx.x * blockDim.x + threadIdx.x;
  if (i >= N * K) return;
  int k = i / N, n = i % N;
  out[i] = (fp16_t)in[n * K + k];
}

// q_in (Nb,64,256) f16 gather from x (B,C,H,W)
__global__ void gather_q_kernel(const float* __restrict__ x, fp16_t* __restrict__ qin) {
  size_t i = (size_t)blockIdx.x * 256 + threadIdx.x;   // 512*64*256 total
  int c  = (int)(i & 255);
  int p  = (int)((i >> 8) & 63);
  int nb = (int)(i >> 14);
  int b = nb >> 6, by = (nb >> 3) & 7, bx = nb & 7;
  int h = by * 8 + (p >> 3), w = bx * 8 + (p & 7);
  qin[i] = (fp16_t)x[(((size_t)b * 256 + c) * 64 + h) * 64 + w];
}

// kv_in (Nb,256,256) f16 gather of R x R halo windows, zero-padded
__global__ void gather_kv_kernel(const float* __restrict__ x, fp16_t* __restrict__ kvin) {
  size_t i = (size_t)blockIdx.x * 256 + threadIdx.x;   // 512*256*256 total
  int c  = (int)(i & 255);
  int r  = (int)((i >> 8) & 255);
  int nb = (int)(i >> 16);
  int b = nb >> 6, by = (nb >> 3) & 7, bx = nb & 7;
  int h = by * 8 - 4 + (r >> 4);
  int w = bx * 8 - 4 + (r & 15);
  float v = 0.0f;
  if ((unsigned)h < 64u && (unsigned)w < 64u)
    v = x[(((size_t)b * 256 + c) * 64 + h) * 64 + w];
  kvin[i] = (fp16_t)v;
}

// ---------------------------------------------------------------------------
// Generic WMMA GEMM: C(M x N) = A(M x K, f16 row-major) * B(K x N, f16 row-major)
// Tile 128x64, BK=32, 256 threads = 8 waves, one 16-row strip per wave.
// A tiles arrive via double-buffered TDM; B tiles staged transposed by VALU.
// MODE 0: q-projection  -> out_h[m*N+n] = v*scale         (scaled q, f16)
// MODE 1: out-projection-> y16 in (b,h,w,c) layout, + bo  (f16)
// MODE 2: 1x1 conv      -> d_out with fused PixelShuffle, + b_conv (f32)
// ---------------------------------------------------------------------------
template <int MODE>
__global__ __launch_bounds__(256) void gemm_wmma_kernel(
    const fp16_t* __restrict__ A, const fp16_t* __restrict__ B,
    int K, int N, const float* __restrict__ bias,
    fp16_t* __restrict__ out_h, float* __restrict__ out_f, float scale) {
  __shared__ fp16_t sA[2][128 * 32];   // double-buffered A tile (TDM target)
  __shared__ fp16_t sBT[64 * 32];      // B tile, transposed: [n][k]

  const int tid  = threadIdx.x;
  const int wave = tid >> 5;
  const int lane = tid & 31;
  const int m0 = blockIdx.x * 128;
  const int n0 = blockIdx.y * 64;

  if (wave == 0)  // prologue: async-load first A tile (128 rows x 32 halves)
    tdm_load_2d_f16(&A[(size_t)m0 * K], &sA[0][0], 32u, 128u, (unsigned)K);

  v8f acc[4] = {};

  for (int k0 = 0; k0 < K; k0 += 32) {
    const int cur = (k0 >> 5) & 1;
    // stage B tile transposed: sBT[n][k]
    {
      int n = tid & 63, kb = (tid >> 6) * 8;
#pragma unroll
      for (int e = 0; e < 8; ++e)
        sBT[n * 32 + kb + e] = B[(size_t)(k0 + kb + e) * N + n0 + n];
    }
    if (k0 + 32 < K)  // gfx1250 global_prefetch_b8 on next B tile
      __builtin_prefetch((const void*)&B[(size_t)(k0 + 32 + (tid >> 3)) * N + n0], 0, 1);
    if (wave == 0) {
      if (k0 + 32 < K) {
        tdm_load_2d_f16(&A[(size_t)m0 * K + k0 + 32], &sA[cur ^ 1][0],
                        32u, 128u, (unsigned)K);
        __builtin_amdgcn_s_wait_tensorcnt(1);   // current tile landed
      } else {
        __builtin_amdgcn_s_wait_tensorcnt(0);
      }
    }
    __syncthreads();

    v16h af = load_a_frag(&sA[cur][wave * 16 * 32], 32);
#pragma unroll
    for (int ct = 0; ct < 4; ++ct) {
      v16h bf = load_b_frag(sBT + ct * 16 * 32, 32);
      acc[ct] = WMMA_F16(af, bf, acc[ct]);
    }
    __syncthreads();
  }

  // epilogue; C/D layout: m = vgpr + 8*(lane>=16), n = lane&15
  const int hf = lane >> 4, nl = lane & 15;
#pragma unroll
  for (int ct = 0; ct < 4; ++ct) {
#pragma unroll
    for (int vr = 0; vr < 8; ++vr) {
      const int m = m0 + wave * 16 + vr + 8 * hf;
      const int n = n0 + ct * 16 + nl;
      const float v = acc[ct][vr];
      if constexpr (MODE == 0) {
        out_h[(size_t)m * N + n] = (fp16_t)(v * scale);
      } else if constexpr (MODE == 1) {
        const int nb = m >> 6, p = m & 63;
        const int b = nb >> 6, by = (nb >> 3) & 7, bx = nb & 7;
        const int h = by * 8 + (p >> 3), w = bx * 8 + (p & 7);
        out_h[(((size_t)(b * 64 + h)) * 64 + w) * 256 + n] = (fp16_t)(v + bias[n]);
      } else {
        const int b = m >> 12, h = (m >> 6) & 63, w = m & 63;
        const int cn = n >> 2, r2 = (n >> 1) & 1, s2 = n & 1;
        out_f[(((size_t)(b * 256 + cn)) * 128 + 2 * h + r2) * 128 + 2 * w + s2] =
            v + bias[n];
      }
    }
  }
}

// ---------------------------------------------------------------------------
// KV projection pass (PASS 0 -> k, PASS 1 -> v), compile-time PASS so the
// LDS write-out is branch-free. A chunks (256x32) via double-buffered TDM.
// ---------------------------------------------------------------------------
template <int PASS>
__device__ __forceinline__ void kv_proj_pass(
    const fp16_t* __restrict__ Akv, const fp16_t* __restrict__ wkvT, int head,
    fp16_t* sA, fp16_t* sBT, fp16_t* sK, fp16_t* sV) {
  const int tid = threadIdx.x, wave = tid >> 5, lane = tid & 31;
  const int hf = lane >> 4, nl = lane & 15;
  const int c0 = head * 64 + PASS * 256;

  if (wave == 0)  // prologue A chunk (rows 0..255, cols 0..31)
    tdm_load_2d_f16(Akv, sA, 32u, 256u, 256u);

  // stage full-K B slice transposed: sBT[n][k] = wkvT[k][c0+n]
  {
    const int n = tid & 63, kb = (tid >> 6) * 64;
    for (int e = 0; e < 64; ++e)
      sBT[n * 256 + kb + e] = wkvT[(size_t)(kb + e) * 512 + c0 + n];
  }

  v8f acc[2][4] = {};
  for (int k0 = 0; k0 < 256; k0 += 32) {
    const int cur = (k0 >> 5) & 1;
    if (wave == 0) {
      if (k0 + 32 < 256) {
        tdm_load_2d_f16(Akv + k0 + 32, sA + (cur ^ 1) * (256 * 32),
                        32u, 256u, 256u);
        __builtin_amdgcn_s_wait_tensorcnt(1);
      } else {
        __builtin_amdgcn_s_wait_tensorcnt(0);
      }
    }
    __syncthreads();
#pragma unroll
    for (int rr = 0; rr < 2; ++rr) {
      v16h af = load_a_frag(sA + cur * (256 * 32) + (wave * 2 + rr) * 16 * 32, 32);
#pragma unroll
      for (int ct = 0; ct < 4; ++ct) {
        v16h bf = load_b_frag(sBT + ct * 16 * 256 + k0, 256);
        acc[rr][ct] = WMMA_F16(af, bf, acc[rr][ct]);
      }
    }
    __syncthreads();
  }

  // write out: PASS 0 -> sK[key][d]; PASS 1 -> sV[d][key] (transposed, packed)
#pragma unroll
  for (int rr = 0; rr < 2; ++rr) {
#pragma unroll
    for (int ct = 0; ct < 4; ++ct) {
      if (PASS == 0) {
#pragma unroll
        for (int vr = 0; vr < 8; ++vr) {
          const int key = (wave * 2 + rr) * 16 + vr + 8 * hf;
          sK[key * 64 + ct * 16 + nl] = (fp16_t)acc[rr][ct][vr];
        }
      } else {
        alignas(16) fp16_t tmp[8];
#pragma unroll
        for (int vr = 0; vr < 8; ++vr) tmp[vr] = (fp16_t)acc[rr][ct][vr];
        const int keyb = (wave * 2 + rr) * 16 + 8 * hf;   // 8 consecutive keys
        const int dd = ct * 16 + nl;
        *(uint4*)&sV[dd * 256 + keyb] = *(const uint4*)tmp;
      }
    }
  }
  __syncthreads();
}

// ---------------------------------------------------------------------------
// Fused halo attention: one workgroup per (window nb, head).
// ---------------------------------------------------------------------------
__global__ __launch_bounds__(256) void halo_attn_kernel(
    const fp16_t* __restrict__ q16,    // (Nb,64,256) pre-scaled by d^-0.5
    const fp16_t* __restrict__ kvin,   // (Nb,256,256)
    const fp16_t* __restrict__ wkvT,   // (256,512) K x N
    const float* __restrict__ rel_h,   // (31,64)
    const float* __restrict__ rel_w,   // (31,64)
    fp16_t* __restrict__ out16) {      // (Nb,64,256)
  extern __shared__ char smem[];
  fp16_t* sA   = (fp16_t*)smem;            // 2 x 256*32 (32 KB) TDM A staging
  fp16_t* sBT  = sA + 2 * 256 * 32;        // 64*256     (32 KB) wkv slice [n][k]
  fp16_t* sK   = sBT + 64 * 256;           // 256*64     (32 KB) k: [key][d]
  fp16_t* sV   = sK + 256 * 64;            // 64*256     (32 KB) v^T: [d][key]
  float*  sSim = (float*)(sV + 64 * 256);  // 64*256     (64 KB)
  float*  sLW  = sSim + 64 * 256;          // 64*16
  float*  sLH  = sLW + 64 * 16;            // 64*16
  fp16_t* sAttn = sK;                      // reuse k buffer after sim

  const int tid = threadIdx.x, wave = tid >> 5, lane = tid & 31;
  const int hf = lane >> 4, nl = lane & 15;
  const int nb = blockIdx.x >> 2, head = blockIdx.x & 3;
  const fp16_t* Akv  = kvin + (size_t)nb * 256 * 256;
  const fp16_t* qptr = q16 + (size_t)nb * 64 * 256 + head * 64;

  kv_proj_pass<0>(Akv, wkvT, head, sA, sBT, sK, sV);
  kv_proj_pass<1>(Akv, wkvT, head, sA, sBT, sK, sV);

  // ---- sim = q @ k^T : 64x256, wave -> row-tile (w&3), col-tiles 8*(w>>2).. ----
  {
    const int rt = wave & 3, cb = (wave >> 2) * 8;
    v8f acc[8] = {};
#pragma unroll
    for (int kk0 = 0; kk0 < 64; kk0 += 32) {
      v16h af = load_a_frag(qptr + rt * 16 * 256 + kk0, 256);  // from global
#pragma unroll
      for (int j = 0; j < 8; ++j) {
        v16h bf = load_b_frag(sK + (cb + j) * 16 * 64 + kk0, 64);
        acc[j] = WMMA_F16(af, bf, acc[j]);
      }
    }
#pragma unroll
    for (int j = 0; j < 8; ++j)
#pragma unroll
      for (int vr = 0; vr < 8; ++vr) {
        const int p   = rt * 16 + vr + 8 * hf;
        const int key = (cb + j) * 16 + nl;
        sSim[p * 256 + key] = acc[j][vr];
      }
  }
  __syncthreads();

  // ---- relative position logit tables: lw[p][j], lh[p][i] ----
  for (int id = tid; id < 2048; id += 256) {
    const int which = id >> 10;        // 0 -> lw (rel_w), 1 -> lh (rel_h)
    const int rem = id & 1023;
    const int p = rem >> 4, o = rem & 15;
    const int a = which ? (p >> 3) : (p & 7);
    const float* tbl = which ? rel_h : rel_w;
    const int ridx = o - a + 15;
    float s = 0.0f;
    for (int kk = 0; kk < 64; ++kk)
      s += (float)qptr[p * 256 + kk] * tbl[ridx * 64 + kk];
    (which ? sLH : sLW)[p * 16 + o] = s;
  }
  __syncthreads();

  // ---- add rel logits + halo mask ----
  {
    const int by = (nb >> 3) & 7, bx = nb & 7;
    for (int i = tid; i < 64 * 256; i += 256) {
      const int p = i >> 8, key = i & 255;
      const int ki = key >> 4, kj = key & 15;
      const int h = by * 8 - 4 + ki, w = bx * 8 - 4 + kj;
      float v = sSim[i] + sLW[p * 16 + kj] + sLH[p * 16 + ki];
      if ((unsigned)h >= 64u || (unsigned)w >= 64u) v = -3.0e38f;
      sSim[i] = v;
    }
  }
  __syncthreads();

  // ---- softmax over 256 keys per row; write attn (f16) into reused buffer ----
  if (tid < 64) {
    const int p = tid;
    float mx = -3.4e38f;
    for (int key = 0; key < 256; ++key) mx = fmaxf(mx, sSim[p * 256 + key]);
    float sum = 0.0f;
    for (int key = 0; key < 256; ++key) {
      float e = __expf(sSim[p * 256 + key] - mx);
      sSim[p * 256 + key] = e;
      sum += e;
    }
    const float inv = 1.0f / sum;
    for (int key = 0; key < 256; ++key)
      sAttn[p * 256 + key] = (fp16_t)(sSim[p * 256 + key] * inv);
  }
  __syncthreads();

  // ---- out = attn @ v : 64x64, wave -> (rt = w>>1, ct pair = (w&1)*2) ----
  {
    const int rt = wave >> 1, cb = (wave & 1) * 2;
    v8f acc[2] = {};
    for (int k0 = 0; k0 < 256; k0 += 32) {
      v16h af = load_a_frag(sAttn + rt * 16 * 256 + k0, 256);
#pragma unroll
      for (int j = 0; j < 2; ++j) {
        v16h bf = load_b_frag(sV + (cb + j) * 16 * 256 + k0, 256);
        acc[j] = WMMA_F16(af, bf, acc[j]);
      }
    }
    fp16_t* op = out16 + (size_t)nb * 64 * 256 + head * 64;
#pragma unroll
    for (int j = 0; j < 2; ++j)
#pragma unroll
      for (int vr = 0; vr < 8; ++vr) {
        const int p  = rt * 16 + vr + 8 * hf;
        const int dd = (cb + j) * 16 + nl;
        op[(size_t)p * 256 + dd] = (fp16_t)acc[j][vr];
      }
  }
}

// ---------------------------------------------------------------------------
extern "C" void kernel_launch(void* const* d_in, const int* in_sizes, int n_in,
                              void* d_out, int out_size, void* d_ws, size_t ws_size,
                              hipStream_t stream) {
  const float* x     = (const float*)d_in[0];
  const float* wq    = (const float*)d_in[1];
  const float* wkv   = (const float*)d_in[2];
  const float* wo    = (const float*)d_in[3];
  const float* bo    = (const float*)d_in[4];
  const float* relh  = (const float*)d_in[5];
  const float* relw  = (const float*)d_in[6];
  const float* wconv = (const float*)d_in[7];
  const float* bconv = (const float*)d_in[8];
  float* out = (float*)d_out;

  // workspace partition (~135 MB, all 16B aligned)
  char* p = (char*)d_ws;
  fp16_t* wqT    = (fp16_t*)p; p += (size_t)256 * 256 * 2;
  fp16_t* wkvT   = (fp16_t*)p; p += (size_t)256 * 512 * 2;
  fp16_t* woT    = (fp16_t*)p; p += (size_t)256 * 256 * 2;
  fp16_t* wconvT = (fp16_t*)p; p += (size_t)256 * 1024 * 2;
  fp16_t* qin    = (fp16_t*)p; p += (size_t)512 * 64 * 256 * 2;
  fp16_t* kvin   = (fp16_t*)p; p += (size_t)512 * 256 * 256 * 2;
  fp16_t* q16    = (fp16_t*)p; p += (size_t)512 * 64 * 256 * 2;
  fp16_t* out16  = (fp16_t*)p; p += (size_t)512 * 64 * 256 * 2;
  fp16_t* y16    = (fp16_t*)p; p += (size_t)8 * 64 * 64 * 256 * 2;

  transpose_w_kernel<<<256,  256, 0, stream>>>(wq,    wqT,    256,  256);
  transpose_w_kernel<<<512,  256, 0, stream>>>(wkv,   wkvT,   512,  256);
  transpose_w_kernel<<<256,  256, 0, stream>>>(wo,    woT,    256,  256);
  transpose_w_kernel<<<1024, 256, 0, stream>>>(wconv, wconvT, 1024, 256);

  gather_q_kernel<<<32768,  256, 0, stream>>>(x, qin);
  gather_kv_kernel<<<131072, 256, 0, stream>>>(x, kvin);

  // Q projection (scaled by d^-0.5 = 0.125): M=32768, K=256, N=256
  gemm_wmma_kernel<0><<<dim3(256, 4), 256, 0, stream>>>(
      qin, wqT, 256, 256, nullptr, q16, nullptr, 0.125f);

  // fused halo attention, one WG per (window, head); 200 KB dynamic LDS
  const size_t smem_bytes =
      (size_t)(2 * 256 * 32 + 64 * 256 + 256 * 64 + 64 * 256) * 2 +
      (size_t)(64 * 256 + 2 * 64 * 16) * 4;
  halo_attn_kernel<<<2048, 256, smem_bytes, stream>>>(
      q16, kvin, wkvT, relh, relw, out16);

  // out projection + bias -> y16 in (b,h,w,c): M=32768, K=256, N=256
  gemm_wmma_kernel<1><<<dim3(256, 4), 256, 0, stream>>>(
      out16, woT, 256, 256, bo, y16, nullptr, 1.0f);

  // 1x1 conv + bias + fused PixelShuffle(2) -> d_out: M=32768, K=256, N=1024
  gemm_wmma_kernel<2><<<dim3(256, 16), 256, 0, stream>>>(
      y16, wconvT, 256, 1024, bconv, nullptr, out, 1.0f);
}
